// MC_E_GCL_19791209300069
// MI455X (gfx1250) — matrix operand order repaired and checked by reading, not compiled, not run on hardware
//
#include <hip/hip_runtime.h>

typedef __attribute__((ext_vector_type(16))) _Float16 v16h;
typedef __attribute__((ext_vector_type(8)))  float    v8f;

#define NNODE 10000
#define NEDGE 320000
#define FDIM  128
#define EIN   265
#define EINP  288     // K padded to 9*32
#define EHID  1060
#define EHIDP 1072    // N padded to 67*16
#define EHIDK 1088    // K padded to 34*32
#define NIN   256
#define NHID  1024
#define CHID  512
#define MT    32      // edges per block (two 16-row WMMA M-tiles)

// ---- workspace layout (float offsets) ----
#define WS_RNORM 0            // 16 floats (use 9)
#define WS_CNT   16           // NNODE
#define WS_SUMS  10016        // NNODE*9
#define WS_MSG   100016       // NNODE*FDIM
#define WS_WPACK 1380016      // f16 packed weights start (32B aligned)
// packed-weight half offsets
#define P_W1   0
#define P_W1_SZ  (67*9*512)
#define P_W2   (P_W1 + P_W1_SZ)
#define P_W2_SZ  (8*34*512)
#define P_CW1  (P_W2 + P_W2_SZ)
#define P_CW1_SZ (32*4*512)
#define P_NW1  (P_CW1 + P_CW1_SZ)
#define P_NW1_SZ (64*8*512)
#define P_NW2  (P_NW1 + P_NW1_SZ)
#define P_NW2_SZ (8*32*512)

__device__ __forceinline__ float siluf(float x) { return x / (1.0f + __expf(-x)); }

__device__ __forceinline__ float wave_sum(float v) {
#pragma unroll
  for (int m = 16; m > 0; m >>= 1) v += __shfl_xor(v, m, 32);
  return v;
}

__device__ __forceinline__ v8f wmma16(v16h a, v16h b, v8f c) {
  return __builtin_amdgcn_wmma_f32_16x16x32_f16(false, a, false, b, (short)0, c,
                                                false, false);
}

// A fragment (16x32 f16) from LDS, row-major [16][K] halves. ISA 7.12.2 layout:
// lane L: M=L%16; lanes 0-15 start K at 0, lanes 16-31 at 8; dwords 0-3 cover
// K pairs {kb..kb+7}, dwords 4-7 cover {kb+16..kb+23}.
__device__ __forceinline__ v16h load_a_frag(const unsigned int* A32, int strideU32,
                                            int lane, int c) {
  int M = lane & 15;
  int base = M * strideU32 + c * 16 + ((lane & 16) ? 4 : 0);
  union { v16h h; unsigned int u[8]; } r;
#pragma unroll
  for (int v = 0; v < 4; ++v) r.u[v] = A32[base + v];
#pragma unroll
  for (int v = 0; v < 4; ++v) r.u[4 + v] = A32[base + 8 + v];
  return r.h;
}

// B fragment: prepacked so each lane reads 16 contiguous halves (32 bytes).
__device__ __forceinline__ v16h load_b_frag(const _Float16* P, long fragIdx) {
  return *(const v16h*)(P + fragIdx * 16);
}

// ---------------- init: zero accumulators ----------------
__global__ void egcl_init_kernel(float* ws) {
  long tot = WS_MSG + (long)NNODE * FDIM;
  for (long i = (long)blockIdx.x * blockDim.x + threadIdx.x; i < tot;
       i += (long)gridDim.x * blockDim.x)
    ws[i] = 0.0f;
}

// ---------------- pack W[K][N] f32 -> B-fragment-ordered f16 ----------------
__global__ void egcl_pack_kernel(const float* __restrict__ W, _Float16* __restrict__ P,
                                 int K, int Nn, int KC, int NT) {
  long tot = (long)NT * KC * 512;
  long i = (long)blockIdx.x * blockDim.x + threadIdx.x;
  if (i >= tot) return;
  int hh = (int)(i & 15);
  int lane = (int)((i >> 4) & 31);
  long rest = i >> 9;
  int c = (int)(rest % KC);
  int t = (int)(rest / KC);
  int n = t * 16 + (lane & 15);
  int k = c * 32 + ((lane & 16) ? 8 : 0) + (hh & 7) + ((hh & 8) ? 16 : 0);
  float v = (k < K && n < Nn) ? W[(long)k * Nn + n] : 0.0f;
  P[i] = (_Float16)v;
}

// ---------------- pass 1: column norms of radial over E ----------------
__global__ void egcl_radial_norm_kernel(const int* __restrict__ ei,
                                        const float* __restrict__ coord,
                                        float* __restrict__ rnorm2) {
  __shared__ float bsum[9];
  int t = threadIdx.x;
  if (t < 9) bsum[t] = 0.0f;
  __syncthreads();
  long e = (long)blockIdx.x * blockDim.x + t;
  if (e < NEDGE) {
    int r = ei[e], c = ei[NEDGE + e];
    float cd[9];
#pragma unroll
    for (int j = 0; j < 9; ++j) cd[j] = coord[(long)r * 9 + j] - coord[(long)c * 9 + j];
#pragma unroll
    for (int cc = 0; cc < 3; ++cc)
#pragma unroll
      for (int ff = 0; ff < 3; ++ff) {
        float d = cd[cc * 3 + 0] * cd[ff * 3 + 0] + cd[cc * 3 + 1] * cd[ff * 3 + 1] +
                  cd[cc * 3 + 2] * cd[ff * 3 + 2];
        atomicAdd(&bsum[cc * 3 + ff], d * d);
      }
  }
  __syncthreads();
  if (t < 9) atomicAdd(&rnorm2[t], bsum[t]);
}

// ---------------- edge kernel: edge MLP + coord MLP + scatters ----------------
// MT=32 edges per block: each B fragment load feeds two WMMA M-tiles,
// halving L2 weight-streaming traffic vs MT=16.
__global__ __launch_bounds__(256, 2) void egcl_edge_kernel(
    const float* __restrict__ h, const int* __restrict__ ei,
    const float* __restrict__ coord,
    const float* __restrict__ ewg, const float* __restrict__ ewb,
    const float* __restrict__ ewb1, const float* __restrict__ ewb2,
    const float* __restrict__ cwg, const float* __restrict__ cwb,
    const float* __restrict__ cw2,
    const _Float16* __restrict__ w1p, const _Float16* __restrict__ w2p,
    const _Float16* __restrict__ cw1p, float* __restrict__ ws) {
  // lifetimes: sH hosts sIn f32[32][268] (phase1), hidden f16[32][1088] (GEMM1/2),
  // then sY f16[32][512] (coord). sA hosts LN'ed input f16[32][288], then sCA.
  __shared__ _Float16 sH[MT * EHIDK];    // 69632 B
  __shared__ _Float16 sA[MT * EINP];     // 18432 B
  __shared__ float sEF[MT * FDIM];       // 16384 B
  __shared__ int srow[MT], scol[MT];
  __shared__ float scd[MT * 9], sRad[MT * 9], sInv[16], scw[MT * 3];

  float* sIn = (float*)sH;         // [32][268]
  const int INROW = 268;
  _Float16* sY = sH;               // [32][512]
  _Float16* sCA = sA;              // [32][128]

  float* rnorm2 = ws + WS_RNORM;
  float* cnt = ws + WS_CNT;
  float* sums = ws + WS_SUMS;
  float* msg = ws + WS_MSG;

  int t = threadIdx.x;
  int lane = t & 31, wave = t >> 5;
  long e0 = (long)blockIdx.x * MT;

  // ---- phase 0: indices, coord diffs, inverse norms ----
  if (t < 9) {
    float nrm = sqrtf(rnorm2[t]);
    sInv[t] = 1.0f / fmaxf(nrm, 1e-12f);
  }
  if (t < MT) {
    long e = e0 + t;
    int r = ei[e], c = ei[NEDGE + e];
    srow[t] = r; scol[t] = c;
#pragma unroll
    for (int j = 0; j < 9; ++j)
      scd[t * 9 + j] = coord[(long)r * 9 + j] - coord[(long)c * 9 + j];
  }
  __syncthreads();
  for (int i = t; i < MT * 9; i += 256) {
    int e = i / 9, j = i % 9, cc = j / 3, ff = j % 3;
    float d = scd[e * 9 + cc * 3 + 0] * scd[e * 9 + ff * 3 + 0] +
              scd[e * 9 + cc * 3 + 1] * scd[e * 9 + ff * 3 + 1] +
              scd[e * 9 + cc * 3 + 2] * scd[e * 9 + ff * 3 + 2];
    sRad[i] = d * sInv[j];
  }
  __syncthreads();

  // ---- phase 1: gather edge input [32][265] ----
  for (int i = t; i < MT * EIN; i += 256) {
    int e = i / EIN, k = i - e * EIN;
    float v;
    if (k < FDIM)            v = h[(long)srow[e] * FDIM + k];
    else if (k < 2 * FDIM)   v = h[(long)scol[e] * FDIM + (k - FDIM)];
    else                     v = sRad[e * 9 + (k - 2 * FDIM)];
    sIn[e * INROW + k] = v;
  }
  __syncthreads();

  // ---- phase 2: LayerNorm(265) -> f16 A tiles [32][288] (zero pad) ----
  for (int e = wave; e < MT; e += 8) {
    float s = 0.f, s2 = 0.f;
    for (int k = lane; k < EIN; k += 32) { float v = sIn[e * INROW + k]; s += v; s2 += v * v; }
    s = wave_sum(s); s2 = wave_sum(s2);
    float mean = s / EIN;
    float rstd = rsqrtf(s2 / EIN - mean * mean + 1e-5f);
    for (int k = lane; k < EINP; k += 32) {
      float v = 0.f;
      if (k < EIN) v = (sIn[e * INROW + k] - mean) * rstd * ewg[k] + ewb[k];
      sA[e * EINP + k] = (_Float16)v;
    }
  }
  __syncthreads();

  // zero hidden pad columns [1072,1088)
  for (int i = t; i < MT * 16; i += 256)
    sH[(i >> 4) * EHIDK + EHIDP + (i & 15)] = (_Float16)0.0f;
  __syncthreads();

  // ---- phase 3: GEMM1 [32x288]x[288x1072] -> silu -> sH f16 ----
  {
    const unsigned int* A32 = (const unsigned int*)sA;
    const int strA = EINP / 2;
    for (int tt = wave; tt < 67; tt += 8) {
      v8f acc0 = {}, acc1 = {};
#pragma unroll
      for (int c = 0; c < 9; ++c) {
        v16h b = load_b_frag(w1p, (long)(tt * 9 + c) * 32 + lane);
        v16h a0 = load_a_frag(A32, strA, lane, c);
        v16h a1 = load_a_frag(A32 + 16 * strA, strA, lane, c);
        acc0 = wmma16(a0, b, acc0);
        acc1 = wmma16(a1, b, acc1);
      }
      int n = tt * 16 + (lane & 15);
      float bb = (n < EHID) ? ewb1[n] : 0.0f;
      int mb = (lane & 16) ? 8 : 0;
#pragma unroll
      for (int i = 0; i < 8; ++i) {
        sH[(mb + i) * EHIDK + n]        = (_Float16)siluf(acc0[i] + bb);
        sH[(16 + mb + i) * EHIDK + n]   = (_Float16)siluf(acc1[i] + bb);
      }
    }
  }
  __syncthreads();

  // ---- phase 4: GEMM2 [32x1088]x[1088x128] -> silu -> edge_feat f32 ----
  {
    const unsigned int* H32 = (const unsigned int*)sH;
    const int strH = EHIDK / 2;
    int tt = wave;  // 8 tiles of 16
    v8f acc0 = {}, acc1 = {};
    for (int c = 0; c < 34; ++c) {
      v16h b = load_b_frag(w2p, (long)(tt * 34 + c) * 32 + lane);
      v16h a0 = load_a_frag(H32, strH, lane, c);
      v16h a1 = load_a_frag(H32 + 16 * strH, strH, lane, c);
      acc0 = wmma16(a0, b, acc0);
      acc1 = wmma16(a1, b, acc1);
    }
    int n = tt * 16 + (lane & 15);
    float bb = ewb2[n];
    int mb = (lane & 16) ? 8 : 0;
#pragma unroll
    for (int i = 0; i < 8; ++i) {
      sEF[(mb + i) * FDIM + n]      = siluf(acc0[i] + bb);
      sEF[(16 + mb + i) * FDIM + n] = siluf(acc1[i] + bb);
    }
  }
  __syncthreads();

  // ---- phase 5: scatter msg/cnt; coord LayerNorm -> f16 ----
  for (int i = t; i < MT * FDIM; i += 256) {
    int e = i >> 7, k = i & 127;
    atomicAdd(&msg[(long)srow[e] * FDIM + k], sEF[e * FDIM + k]);
  }
  if (t < MT) atomicAdd(&cnt[srow[t]], 1.0f);
  for (int e = wave; e < MT; e += 8) {
    float s = 0.f, s2 = 0.f;
    for (int k = lane; k < FDIM; k += 32) { float v = sEF[e * FDIM + k]; s += v; s2 += v * v; }
    s = wave_sum(s); s2 = wave_sum(s2);
    float mean = s / FDIM;
    float rstd = rsqrtf(s2 / FDIM - mean * mean + 1e-5f);
    for (int k = lane; k < FDIM; k += 32)
      sCA[e * FDIM + k] = (_Float16)((sEF[e * FDIM + k] - mean) * rstd * cwg[k] + cwb[k]);
  }
  __syncthreads();

  // ---- phase 6: coord GEMM [32x128]x[128x512] -> silu -> sY f16 ----
  {
    const unsigned int* CA32 = (const unsigned int*)sCA;
    const int strC = FDIM / 2;
    for (int tt = wave; tt < 32; tt += 8) {
      v8f acc0 = {}, acc1 = {};
#pragma unroll
      for (int c = 0; c < 4; ++c) {
        v16h b = load_b_frag(cw1p, (long)(tt * 4 + c) * 32 + lane);
        v16h a0 = load_a_frag(CA32, strC, lane, c);
        v16h a1 = load_a_frag(CA32 + 16 * strC, strC, lane, c);
        acc0 = wmma16(a0, b, acc0);
        acc1 = wmma16(a1, b, acc1);
      }
      int n = tt * 16 + (lane & 15);
      int mb = (lane & 16) ? 8 : 0;
#pragma unroll
      for (int i = 0; i < 8; ++i) {
        sY[(mb + i) * CHID + n]      = (_Float16)siluf(acc0[i]);
        sY[(16 + mb + i) * CHID + n] = (_Float16)siluf(acc1[i]);
      }
    }
  }
  __syncthreads();

  // ---- phase 7: [512]->3 projection, weighted coord-diff scatter ----
  if (t < MT * 3) {
    int e = t / 3, c = t % 3;
    float acc = 0.f;
    for (int k = 0; k < CHID; ++k) acc += (float)sY[e * CHID + k] * cw2[k * 3 + c];
    scw[t] = acc;
  }
  __syncthreads();
  for (int i = t; i < MT * 9; i += 256) {
    int e = i / 9, j = i % 9, c = j / 3;
    atomicAdd(&sums[(long)srow[e] * 9 + j], scd[e * 9 + j] * scw[e * 3 + c]);
  }
}

// ---------------- node kernel: node MLP + residual ----------------
__global__ __launch_bounds__(256, 2) void egcl_node_kernel(
    const float* __restrict__ h, const float* __restrict__ msg,
    const float* __restrict__ nwg, const float* __restrict__ nwb,
    const float* __restrict__ nwb1, const float* __restrict__ nwb2,
    const _Float16* __restrict__ nw1p, const _Float16* __restrict__ nw2p,
    float* __restrict__ hout) {
  __shared__ _Float16 sHn[16 * NHID];   // phase1 reused as sIn f32[16][256]
  __shared__ _Float16 sA[16 * NIN];
  float* sIn = (float*)sHn;

  int t = threadIdx.x, lane = t & 31, wave = t >> 5;
  long n0 = (long)blockIdx.x * 16;

  for (int i = t; i < 16 * NIN; i += 256) {
    int e = i >> 8, k = i & 255;
    float v = (k < FDIM) ? h[(n0 + e) * FDIM + k] : msg[(n0 + e) * FDIM + (k - FDIM)];
    sIn[e * NIN + k] = v;
  }
  __syncthreads();
  for (int e = wave; e < 16; e += 8) {
    float s = 0.f, s2 = 0.f;
    for (int k = lane; k < NIN; k += 32) { float v = sIn[e * NIN + k]; s += v; s2 += v * v; }
    s = wave_sum(s); s2 = wave_sum(s2);
    float mean = s / NIN;
    float rstd = rsqrtf(s2 / NIN - mean * mean + 1e-5f);
    for (int k = lane; k < NIN; k += 32)
      sA[e * NIN + k] = (_Float16)((sIn[e * NIN + k] - mean) * rstd * nwg[k] + nwb[k]);
  }
  __syncthreads();

  // GEMM1 [16x256]x[256x1024] -> silu -> sHn
  {
    const unsigned int* A32 = (const unsigned int*)sA;
    for (int tt = wave; tt < 64; tt += 8) {
      v8f acc = {};
#pragma unroll
      for (int c = 0; c < 8; ++c) {
        v16h a = load_a_frag(A32, NIN / 2, lane, c);
        v16h b = load_b_frag(nw1p, (long)(tt * 8 + c) * 32 + lane);
        acc = wmma16(a, b, acc);
      }
      int n = tt * 16 + (lane & 15);
      float bb = nwb1[n];
      int mb = (lane & 16) ? 8 : 0;
#pragma unroll
      for (int i = 0; i < 8; ++i)
        sHn[(mb + i) * NHID + n] = (_Float16)siluf(acc[i] + bb);
    }
  }
  __syncthreads();

  // GEMM2 [16x1024]x[1024x128] -> silu -> +h residual -> global
  {
    const unsigned int* H32 = (const unsigned int*)sHn;
    int tt = wave;
    v8f acc = {};
    for (int c = 0; c < 32; ++c) {
      v16h a = load_a_frag(H32, NHID / 2, lane, c);
      v16h b = load_b_frag(nw2p, (long)(tt * 32 + c) * 32 + lane);
      acc = wmma16(a, b, acc);
    }
    int n = tt * 16 + (lane & 15);
    float bb = nwb2[n];
    int mb = (lane & 16) ? 8 : 0;
#pragma unroll
    for (int i = 0; i < 8; ++i) {
      long node = n0 + mb + i;
      hout[node * FDIM + n] = siluf(acc[i] + bb) + h[node * FDIM + n];
    }
  }
}

// ---------------- coord output: segment mean, clip, add ----------------
__global__ void egcl_coord_kernel(const float* __restrict__ coord,
                                  const float* __restrict__ sums,
                                  const float* __restrict__ cnt,
                                  float* __restrict__ cout) {
  long i = (long)blockIdx.x * blockDim.x + threadIdx.x;
  if (i < (long)NNODE * 9) {
    float c = cnt[i / 9];
    float a = sums[i] / fmaxf(c, 1.0f);
    a = fminf(fmaxf(a, -10.0f), 10.0f);
    cout[i] = coord[i] + a;
  }
}

extern "C" void kernel_launch(void* const* d_in, const int* in_sizes, int n_in,
                              void* d_out, int out_size, void* d_ws, size_t ws_size,
                              hipStream_t stream) {
  const float* h       = (const float*)d_in[0];
  const int*   ei      = (const int*)d_in[1];
  const float* coord   = (const float*)d_in[2];
  const float* ew_ln_g = (const float*)d_in[3];
  const float* ew_ln_b = (const float*)d_in[4];
  const float* ew_w1   = (const float*)d_in[5];
  const float* ew_b1   = (const float*)d_in[6];
  const float* ew_w2   = (const float*)d_in[7];
  const float* ew_b2   = (const float*)d_in[8];
  const float* nw_ln_g = (const float*)d_in[9];
  const float* nw_ln_b = (const float*)d_in[10];
  const float* nw_w1   = (const float*)d_in[11];
  const float* nw_b1   = (const float*)d_in[12];
  const float* nw_w2   = (const float*)d_in[13];
  const float* nw_b2   = (const float*)d_in[14];
  const float* cw_ln_g = (const float*)d_in[15];
  const float* cw_ln_b = (const float*)d_in[16];
  const float* cw_w1   = (const float*)d_in[17];
  const float* cw_w2   = (const float*)d_in[18];

  float* ws = (float*)d_ws;
  _Float16* wp = (_Float16*)(ws + WS_WPACK);
  float* hout = (float*)d_out;
  float* cout = hout + (long)NNODE * FDIM;

  egcl_init_kernel<<<2048, 256, 0, stream>>>(ws);

  egcl_pack_kernel<<<(67 * 9 * 512 + 255) / 256, 256, 0, stream>>>(ew_w1, wp + P_W1,
                                                                   EIN, EHID, 9, 67);
  egcl_pack_kernel<<<(8 * 34 * 512 + 255) / 256, 256, 0, stream>>>(ew_w2, wp + P_W2,
                                                                   EHID, FDIM, 34, 8);
  egcl_pack_kernel<<<(32 * 4 * 512 + 255) / 256, 256, 0, stream>>>(cw_w1, wp + P_CW1,
                                                                   FDIM, CHID, 4, 32);
  egcl_pack_kernel<<<(64 * 8 * 512 + 255) / 256, 256, 0, stream>>>(nw_w1, wp + P_NW1,
                                                                   NIN, NHID, 8, 64);
  egcl_pack_kernel<<<(8 * 32 * 512 + 255) / 256, 256, 0, stream>>>(nw_w2, wp + P_NW2,
                                                                   NHID, FDIM, 32, 8);

  egcl_radial_norm_kernel<<<NEDGE / 256, 256, 0, stream>>>(ei, coord, ws + WS_RNORM);

  egcl_edge_kernel<<<NEDGE / MT, 256, 0, stream>>>(
      h, ei, coord, ew_ln_g, ew_ln_b, ew_b1, ew_b2, cw_ln_g, cw_ln_b, cw_w2,
      wp + P_W1, wp + P_W2, wp + P_CW1, ws);

  egcl_node_kernel<<<NNODE / 16, 256, 0, stream>>>(
      h, ws + WS_MSG, nw_ln_g, nw_ln_b, nw_b1, nw_b2, wp + P_NW1, wp + P_NW2, hout);

  egcl_coord_kernel<<<(NNODE * 9 + 255) / 256, 256, 0, stream>>>(
      coord, ws + WS_SUMS, ws + WS_CNT, cout);
}
